// Product2Vec_62302795595887
// MI455X (gfx1250) — matrix operand order repaired and checked by reading, not compiled, not run on hardware
//
#include <hip/hip_runtime.h>
#include <hip/hip_bf16.h>
#include <math.h>

// Problem constants from the reference
#define B_SZ   2048
#define N_NBR  128
#define D_DIM  512
#define HID    2048
#define H_HEAD 8
#define DH     64
#define EPS_BN 1e-5f

typedef __attribute__((ext_vector_type(16))) __bf16 v16bf;
typedef __attribute__((ext_vector_type(8)))  float  v8f;

// fp32 -> bf16 via the HIP conversion API: lowers to native v_cvt when the
// target supports it, correct software RNE sequence otherwise.
__device__ __forceinline__ unsigned short f2bf(float x) {
  __hip_bfloat16 b = __float2bfloat16(x);
  unsigned short u;
  __builtin_memcpy(&u, &b, sizeof(u));
  return u;
}
// Pack two fp32 -> bf16x2 in one u32 (low = lo, high = hi).
__device__ __forceinline__ unsigned int f2bf_pk(float lo, float hi) {
  __hip_bfloat162 h = __float22bfloat162_rn(make_float2(lo, hi));
  unsigned int u;
  __builtin_memcpy(&u, &h, sizeof(u));
  return u;
}
__device__ __forceinline__ float bf2f(unsigned short h) {
  return __uint_as_float(((unsigned int)h) << 16);
}

// A operand (16x32 bf16, ISA 7.12.2): lane group 0 holds K {0..7, 16..23},
// lane group 1 holds K {8..15, 24..31} of row (lane&15).
// p = &As[row*stride + (lane>>4)*8]; reads at +0 and +16 halves (16B aligned).
__device__ __forceinline__ v16bf load_fragA(const unsigned short* p) {
  union { uint4 q[2]; v16bf v; } u;
  u.q[0] = *(const uint4*)(p);
  u.q[1] = *(const uint4*)(p + 16);
  return u.v;
}
// B operand (32x16 bf16): same K order as A: lane group 0 holds K {0..7, 16..23},
// lane group 1 holds K {8..15, 24..31} of
// column (lane&15).  p = &Bs[col*stride + (lane>>4)*8]; reads at +0 and +16 halves.
__device__ __forceinline__ v16bf load_fragB(const unsigned short* p) {
  union { uint4 q[2]; v16bf v; } u;
  u.q[0] = *(const uint4*)(p);
  u.q[1] = *(const uint4*)(p + 16);
  return u.v;
}

// -------------------------------------------------------------------------
// Generic tiled GEMM:  C[M,N] = epilogue( A[M,K] @ W[K,N] + bias[N] )
// fp32 in memory -> bf16 staged in LDS -> v_wmma_f32_16x16x32_bf16 (fp32 acc).
// Block tile 128x128, BK=32, 256 threads = 8 waves (4x2), each wave 2x4
// WMMA tiles.  Double-buffered LDS + register prefetch: one barrier per
// K-step; next tile's global loads issue before the WMMA block and the
// bf16 conversion/staging co-executes with the matrix pipe.
// EPI: 0 = identity, 1 = tanh.  NORM_A: per-K-column affine on A (folded
// BatchNorm; scale/shift precomputed).  OUT_BF16: store bf16 output.
// -------------------------------------------------------------------------
template <int EPI, bool NORM_A, bool OUT_BF16>
__global__ __launch_bounds__(256)
void gemm_wmma_bf16(const float* __restrict__ A, const float* __restrict__ W,
                    const float* __restrict__ bias,
                    const float* __restrict__ scale,
                    const float* __restrict__ shift,
                    void* __restrict__ Cout, int M, int N, int K) {
  constexpr int LDSS = 48;  // LDS row stride in halves (96B; keeps 16B align)
  __shared__ __align__(16) unsigned short As[2][128 * LDSS];
  __shared__ __align__(16) unsigned short Bs[2][128 * LDSS];

  const int t    = threadIdx.x;
  const int m0   = blockIdx.y * 128;
  const int n0   = blockIdx.x * 128;
  const int wv   = t >> 5;
  const int lane = t & 31;
  const int wm   = wv >> 1;     // 0..3 -> 32-row stripe
  const int wn   = wv & 1;      // 0..1 -> 64-col stripe
  const int r16  = lane & 15;
  const int hi   = lane >> 4;

  // staging assignments
  const int ar = t >> 3;          // A: rows ar, ar+32, ar+64, ar+96
  const int ac = (t & 7) << 2;    // A: K cols ac..ac+3
  const int br = t >> 5;          // B: K rows br, br+8, br+16, br+24
  const int bc = (t & 31) << 2;   // B: N cols bc..bc+3

  const float* Abase = A + (size_t)m0 * K;
  const float* Wbase = W + n0;

  float4 pa[4], pb[4];
  auto loadA = [&](int k0) {
#pragma unroll
    for (int i = 0; i < 4; ++i) {
      float4 v = *(const float4*)(Abase + (size_t)(ar + (i << 5)) * K + k0 + ac);
      if (NORM_A) {
        const int kg = k0 + ac;
        v.x = v.x * scale[kg + 0] + shift[kg + 0];
        v.y = v.y * scale[kg + 1] + shift[kg + 1];
        v.z = v.z * scale[kg + 2] + shift[kg + 2];
        v.w = v.w * scale[kg + 3] + shift[kg + 3];
      }
      pa[i] = v;
    }
  };
  auto loadB = [&](int k0) {
#pragma unroll
    for (int i = 0; i < 4; ++i)
      pb[i] = *(const float4*)(Wbase + (size_t)(k0 + br + (i << 3)) * N + bc);
  };
  auto stage = [&](int buf) {
#pragma unroll
    for (int i = 0; i < 4; ++i) {
      uint2 apk;
      apk.x = f2bf_pk(pa[i].x, pa[i].y);
      apk.y = f2bf_pk(pa[i].z, pa[i].w);
      *(uint2*)&As[buf][(ar + (i << 5)) * LDSS + ac] = apk;  // one b64 store
      const int k = br + (i << 3);
      Bs[buf][(bc + 0) * LDSS + k] = f2bf(pb[i].x);
      Bs[buf][(bc + 1) * LDSS + k] = f2bf(pb[i].y);
      Bs[buf][(bc + 2) * LDSS + k] = f2bf(pb[i].z);
      Bs[buf][(bc + 3) * LDSS + k] = f2bf(pb[i].w);
    }
  };

  const v8f vzero = {0.f, 0.f, 0.f, 0.f, 0.f, 0.f, 0.f, 0.f};
  v8f acc[2][4];
#pragma unroll
  for (int tm = 0; tm < 2; ++tm)
#pragma unroll
    for (int tn = 0; tn < 4; ++tn) acc[tm][tn] = vzero;

  const int KT = K >> 5;
  loadA(0);
  loadB(0);
  stage(0);
  for (int kt = 0; kt < KT; ++kt) {
    __syncthreads();  // staged buffer (kt&1) now visible to all waves
    const int cur = kt & 1;
    const bool more = (kt + 1 < KT);
    if (more) {  // issue next tile's global loads before the math
      loadA((kt + 1) << 5);
      loadB((kt + 1) << 5);
    }
    v16bf af[2], bfv[4];
#pragma unroll
    for (int tm = 0; tm < 2; ++tm)
      af[tm] = load_fragA(&As[cur][(wm * 32 + tm * 16 + r16) * LDSS + hi * 8]);
#pragma unroll
    for (int tn = 0; tn < 4; ++tn)
      bfv[tn] = load_fragB(&Bs[cur][(wn * 64 + tn * 16 + r16) * LDSS + hi * 8]);
#pragma unroll
    for (int tm = 0; tm < 2; ++tm)
#pragma unroll
      for (int tn = 0; tn < 4; ++tn)
        acc[tm][tn] = __builtin_amdgcn_wmma_f32_16x16x32_bf16(
            false, af[tm], false, bfv[tn], (short)0, acc[tm][tn], false, false);
    if (more) stage(cur ^ 1);  // write other buffer; barrier above covered
                               // its last readers (iteration kt-1)
  }

  // ---- epilogue: C layout — VGPR r, lane: n = lane&15, m = 8*(lane>>4)+r
#pragma unroll
  for (int tm = 0; tm < 2; ++tm) {
#pragma unroll
    for (int tn = 0; tn < 4; ++tn) {
      const int nn = n0 + wn * 64 + tn * 16 + r16;
      const float bi = bias[nn];
#pragma unroll
      for (int r = 0; r < 8; ++r) {
        const int mm = m0 + wm * 32 + tm * 16 + hi * 8 + r;
        float v = acc[tm][tn][r] + bi;
        if (EPI == 1) v = tanhf(v);
        if (OUT_BF16)
          ((unsigned short*)Cout)[(size_t)mm * N + nn] = f2bf(v);
        else
          ((float*)Cout)[(size_t)mm * N + nn] = v;
      }
    }
  }
}

// -------------------------------------------------------------------------
// BatchNorm batch stats folded to per-column affine:
//   scale[c] = gamma[c] * rsqrt(var[c] + eps); shift[c] = beta[c] - mu[c]*scale
// One thread per column; row loads are coalesced across the block.
// -------------------------------------------------------------------------
__global__ __launch_bounds__(256)
void bn_stats(const float* __restrict__ X, const float* __restrict__ gamma,
              const float* __restrict__ beta, float* __restrict__ scale,
              float* __restrict__ shift, int M, int N) {
  const int c = blockIdx.x * 256 + threadIdx.x;
  float s = 0.f, ss = 0.f;
  for (int r = 0; r < M; ++r) {
    const float v = X[(size_t)r * N + c];
    s += v;
    ss += v * v;
  }
  const float inv = 1.0f / (float)M;
  const float m = s * inv;
  const float var = ss * inv - m * m;
  const float sc = gamma[c] * rsqrtf(var + EPS_BN);
  scale[c] = sc;
  shift[c] = beta[c] - m * sc;
}

// -------------------------------------------------------------------------
// Masked single-query multi-head attention.
// Block = 256 threads = 8 waves; block b handles batch item b, wave = head.
// Lane owns neighbors {lane, lane+32, lane+64, lane+96} for scores, then
// head-dims {2*lane, 2*lane+1} for the context accumulation.
// -------------------------------------------------------------------------
__global__ __launch_bounds__(256)
void attn_kernel(const float* __restrict__ qbuf,
                 const unsigned short* __restrict__ kbuf,
                 const unsigned short* __restrict__ vbuf,
                 const int* __restrict__ lengths, float* __restrict__ ctx) {
  const int b = blockIdx.x;
  const int h = threadIdx.x >> 5;
  const int lane = threadIdx.x & 31;
  const int len = lengths[b];

  // Broadcast-load q[b,h,:] into registers (64 floats).
  const float* q = qbuf + (size_t)b * D_DIM + h * DH;
  float qreg[DH];
#pragma unroll
  for (int d = 0; d < DH; ++d) qreg[d] = q[d];

  // ---- scores (scaled, masked)
  float s[4];
#pragma unroll
  for (int j = 0; j < 4; ++j) {
    const int n = lane + 32 * j;
    const unsigned short* kp =
        kbuf + ((size_t)b * N_NBR + n) * D_DIM + h * DH;
    float a = 0.f;
#pragma unroll
    for (int d = 0; d < DH; ++d) a += qreg[d] * bf2f(kp[d]);
    s[j] = (n < len) ? a * 0.125f : -INFINITY;  // 1/sqrt(64)
  }

  // ---- softmax over 128 (wave32 shuffle reductions)
  float m = fmaxf(fmaxf(s[0], s[1]), fmaxf(s[2], s[3]));
#pragma unroll
  for (int off = 16; off > 0; off >>= 1) m = fmaxf(m, __shfl_xor(m, off, 32));
  float e[4];
  float sum = 0.f;
#pragma unroll
  for (int j = 0; j < 4; ++j) {
    e[j] = (s[j] == -INFINITY) ? 0.f : __expf(s[j] - m);
    sum += e[j];
  }
#pragma unroll
  for (int off = 16; off > 0; off >>= 1) sum += __shfl_xor(sum, off, 32);
  const float inv = 1.0f / sum;
#pragma unroll
  for (int j = 0; j < 4; ++j) e[j] *= inv;

  // ---- ctx[d] = sum_n attn[n] * v[b,n,h,d]; lane owns d0, d0+1
  const int d0 = lane * 2;
  float c0 = 0.f, c1 = 0.f;
#pragma unroll
  for (int j = 0; j < 4; ++j) {
    const float aj = e[j];
    for (int src = 0; src < 32; ++src) {
      const float a = __shfl(aj, src, 32);
      const int n = j * 32 + src;
      const unsigned short* vp =
          vbuf + ((size_t)b * N_NBR + n) * D_DIM + h * DH + d0;
      c0 += a * bf2f(vp[0]);
      c1 += a * bf2f(vp[1]);
    }
  }
  float* cp = ctx + (size_t)b * D_DIM + h * DH + d0;
  cp[0] = c0;
  cp[1] = c1;
}

// -------------------------------------------------------------------------
extern "C" void kernel_launch(void* const* d_in, const int* in_sizes, int n_in,
                              void* d_out, int out_size, void* d_ws,
                              size_t ws_size, hipStream_t stream) {
  (void)in_sizes; (void)n_in; (void)out_size; (void)ws_size;
  const float* catalog   = (const float*)d_in[0];
  const float* neighbors = (const float*)d_in[1];  // [B*N, D] row-major
  const int*   lengths   = (const int*)d_in[2];
  const float* W1 = (const float*)d_in[3];
  const float* b1 = (const float*)d_in[4];
  const float* gamma = (const float*)d_in[5];
  const float* beta  = (const float*)d_in[6];
  const float* W2 = (const float*)d_in[7];
  const float* b2 = (const float*)d_in[8];
  const float* W3 = (const float*)d_in[9];
  const float* b3 = (const float*)d_in[10];
  const float* Wq = (const float*)d_in[11];
  const float* bq = (const float*)d_in[12];
  const float* Wk = (const float*)d_in[13];
  const float* bk = (const float*)d_in[14];
  const float* Wv = (const float*)d_in[15];
  const float* bv = (const float*)d_in[16];
  const float* Wo = (const float*)d_in[17];
  const float* bo = (const float*)d_in[18];

  // Workspace carve-up (fp32 intermediates, then bf16 K/V ~256MB each)
  float* ws    = (float*)d_ws;
  float* x1    = ws;                                  // [B, HID]
  float* x2    = x1 + (size_t)B_SZ * HID;             // [B, HID]
  float* scl   = x2 + (size_t)B_SZ * HID;             // [HID]
  float* shf   = scl + HID;                           // [HID]
  float* emb   = shf + HID;                           // [B, D]
  float* qbuf  = emb + (size_t)B_SZ * D_DIM;          // [B, D]
  float* ctx   = qbuf + (size_t)B_SZ * D_DIM;         // [B, D]
  unsigned short* kbuf = (unsigned short*)(ctx + (size_t)B_SZ * D_DIM);
  unsigned short* vbuf = kbuf + (size_t)B_SZ * N_NBR * D_DIM;

  const dim3 blk(256);
  const int MN = B_SZ * N_NBR;  // 262144 neighbor rows

  // 1) X1 = tanh(catalog @ W1 + b1)          [2048,512]x[512,2048]
  gemm_wmma_bf16<1, false, false>
      <<<dim3(HID / 128, B_SZ / 128), blk, 0, stream>>>(
          catalog, W1, b1, nullptr, nullptr, x1, B_SZ, HID, D_DIM);
  // 2) BatchNorm batch stats -> per-column affine (scale, shift)
  bn_stats<<<dim3(HID / 256), blk, 0, stream>>>(x1, gamma, beta, scl, shf,
                                                B_SZ, HID);
  // 3) X2 = tanh(norm(X1) @ W2 + b2)         [2048,2048]x[2048,2048]
  gemm_wmma_bf16<1, true, false>
      <<<dim3(HID / 128, B_SZ / 128), blk, 0, stream>>>(
          x1, W2, b2, scl, shf, x2, B_SZ, HID, HID);
  // 4) emb = X2 @ W3 + b3                    [2048,2048]x[2048,512]
  gemm_wmma_bf16<0, false, false>
      <<<dim3(D_DIM / 128, B_SZ / 128), blk, 0, stream>>>(
          x2, W3, b3, nullptr, nullptr, emb, B_SZ, D_DIM, HID);
  // 5) q = emb @ Wq + bq                     [2048,512]x[512,512]
  gemm_wmma_bf16<0, false, false>
      <<<dim3(D_DIM / 128, B_SZ / 128), blk, 0, stream>>>(
          emb, Wq, bq, nullptr, nullptr, qbuf, B_SZ, D_DIM, D_DIM);
  // 6) k = neighbors @ Wk + bk  (bf16 out)   [262144,512]x[512,512]
  gemm_wmma_bf16<0, false, true>
      <<<dim3(D_DIM / 128, MN / 128), blk, 0, stream>>>(
          neighbors, Wk, bk, nullptr, nullptr, kbuf, MN, D_DIM, D_DIM);
  // 7) v = neighbors @ Wv + bv  (bf16 out)
  gemm_wmma_bf16<0, false, true>
      <<<dim3(D_DIM / 128, MN / 128), blk, 0, stream>>>(
          neighbors, Wv, bv, nullptr, nullptr, vbuf, MN, D_DIM, D_DIM);
  // 8) masked softmax attention -> ctx
  attn_kernel<<<dim3(B_SZ), blk, 0, stream>>>(qbuf, kbuf, vbuf, lengths, ctx);
  // 9) out = ctx @ Wo + bo -> d_out (fp32)
  gemm_wmma_bf16<0, false, false>
      <<<dim3(D_DIM / 128, B_SZ / 128), blk, 0, stream>>>(
          ctx, Wo, bo, nullptr, nullptr, (float*)d_out, B_SZ, D_DIM, D_DIM);
}
